// HDCNN_66108136620367
// MI455X (gfx1250) — matrix operand order repaired and checked by reading, not compiled, not run on hardware
//
#include <hip/hip_runtime.h>
#include <hip/hip_bf16.h>
#include <math.h>

#define INPUT_SIZE 32768
#define FILTER_LEN 8192
#define CONV_LEN   40960      // conv output buffer (160 tiles * 256)
#define NT         257        // K-chunk count per tile (covers l in [-8192, 32))
#define WRP_LEN    8256       // reversed+padded filter length
#define VT_LEN     8464       // per-tile signal window [j0-8192, j0+272)
#define OUT_MAX    65536      // INPUT_SIZE + 4*FILTER_LEN

typedef __attribute__((ext_vector_type(16))) __bf16         v16bf;
typedef __attribute__((ext_vector_type(16))) unsigned short v16u;
typedef __attribute__((ext_vector_type(8)))  unsigned short v8u;
typedef __attribute__((ext_vector_type(8)))  float          v8f;
typedef __attribute__((ext_vector_type(4)))  unsigned int   u32x4;
typedef __attribute__((ext_vector_type(8)))  int            i32x8;
typedef __attribute__((ext_vector_type(4)))  int            i32x4;

#if defined(__has_builtin)
#if __has_builtin(__builtin_amdgcn_tensor_load_to_lds) && \
    __has_builtin(__builtin_amdgcn_s_wait_tensorcnt)
#define HAVE_TDM 1
#endif
#endif

static __device__ __forceinline__ unsigned short f2bf(float f) {
  unsigned u = __float_as_uint(f);
  u += 0x7FFFu + ((u >> 16) & 1u);       // round-to-nearest-even
  return (unsigned short)(u >> 16);
}

// ---------------- bf16 conversion of the running signal ----------------
__global__ void cvt_bf16_kernel(const float* __restrict__ x,
                                unsigned short* __restrict__ y, int n) {
  int i = blockIdx.x * blockDim.x + threadIdx.x;
  if (i < n) y[i] = f2bf(x[i]);
}

// ---------------- reversed + zero-padded filter: wrp[i] = w[8223-i] ----------------
__global__ void build_wrp_kernel(const float* __restrict__ w,
                                 unsigned short* __restrict__ wrp) {
  int i = blockIdx.x * blockDim.x + threadIdx.x;
  if (i < WRP_LEN) {
    int k = 8223 - i;
    wrp[i] = (k >= 0 && k < FILTER_LEN) ? f2bf(w[k]) : (unsigned short)0;
  }
}

// ---------------- B fragments in WMMA 32x16 layout, one per K-chunk ----------------
// B[k][n] = w[n - lbase - k], lbase = -8192 + 32t.  Slot e of lane L holds
// K = e + (L>=16 ? 16 : 0), N = L&15  ->  wrp index 31 - n + dk + 32t + e.
__global__ __launch_bounds__(32) void build_bfrag_kernel(
    const unsigned short* __restrict__ wrp, unsigned short* __restrict__ bfrag) {
  int t = blockIdx.x;                 // 0..256
  int lane = threadIdx.x;             // 0..31
  int n = lane & 15;
  int dk = (lane >> 4) << 4;          // 0 or 16
  int src = 31 - n + dk + 32 * t;
  unsigned short tmp[16];
#pragma unroll
  for (int e = 0; e < 16; ++e) tmp[e] = wrp[src + e];
  v16u vv;
  __builtin_memcpy(&vv, tmp, 32);
  *(v16u*)(bfrag + t * 512 + lane * 16) = vv;   // 32B-aligned
}

// ---------------- one K-chunk: two aligned LDS A-loads + one B-load + WMMA ----------------
static __device__ __forceinline__ v8f conv_step(const unsigned short* ap,
                                                const unsigned short* bp,
                                                int t, v8f acc) {
  // A slots 0..7 = window[base..base+7], slots 8..15 = window[base+8..base+15]
  v8u alo = *(const v8u*)(ap + 32 * t);
  v8u ahi = *(const v8u*)(ap + 32 * t + 8);
  v16u au = __builtin_shufflevector(alo, ahi, 0, 1, 2, 3, 4, 5, 6, 7,
                                    8, 9, 10, 11, 12, 13, 14, 15);
  v16u bu = *(const v16u*)(bp + t * 512);
  v16bf A = __builtin_bit_cast(v16bf, au);
  v16bf B = __builtin_bit_cast(v16bf, bu);
  return __builtin_amdgcn_wmma_f32_16x16x32_bf16(
      false, A, false, B, (short)0, acc, false, false);
}

// ---------------- convolution tile: 256 outputs per wave via 257 WMMAs ----------------
__global__ __launch_bounds__(32) void conv_wmma_kernel(
    const unsigned short* __restrict__ vbf,
    const unsigned short* __restrict__ bfrag,
    float* __restrict__ convout) {
  __shared__ __align__(16) unsigned short vtile[VT_LEN];
  const int lane = threadIdx.x;
  const int j0 = blockIdx.x << 8;
  const int gbase = j0 - FILTER_LEN;

  int lo = gbase < 0 ? 0 : gbase;
  int hi = gbase + VT_LEN;
  if (hi > INPUT_SIZE) hi = INPUT_SIZE;
  const int lpad = lo - gbase;                 // even, >=0 (left zero pad)
  const int rpad = gbase + VT_LEN - hi;        // even, >=0 (right zero pad)

  // zero-fill out-of-range edges (boundary tiles only)
  for (int p = lane; p < (lpad >> 1); p += 32)
    *(unsigned*)&vtile[2 * p] = 0u;
  for (int p = lane; p < (rpad >> 1); p += 32)
    *(unsigned*)&vtile[(hi - gbase) + 2 * p] = 0u;

#if HAVE_TDM
  {
    // 1-D Tensor-DMA descriptor: copy (hi-lo) bf16 elems global -> LDS.
    // vtile is the only __shared__ object -> its LDS byte offset is 0.
    unsigned long long ga = (unsigned long long)(size_t)(vbf + lo);
    unsigned n = (unsigned)(hi - lo);
    u32x4 g0;
    g0[0] = 1u;                                          // count=1 (valid D#)
    g0[1] = (unsigned)(lpad * 2);                        // lds_addr (bytes)
    g0[2] = (unsigned)(ga & 0xFFFFFFFFull);              // global_addr[31:0]
    g0[3] = (unsigned)((ga >> 32) & 0x01FFFFFFull)       // global_addr[56:32]
            | (2u << 30);                                // type=2 ("image")
    i32x8 g1 = {};
    g1[0] = 0x00010000;                                  // data_size=1 -> 2 bytes
    g1[1] = (int)((n & 0xFFFFu) << 16);                  // tensor_dim0[15:0]
    g1[2] = (int)((n >> 16) | (1u << 16));               // tensor_dim0[31:16], tensor_dim1=1
    g1[3] = (int)((n & 0xFFFFu) << 16);                  // tile_dim0 = n
    g1[5] = (int)n;                                      // tensor_dim0_stride[31:0]
    i32x4 gz = {};
#if __clang_major__ >= 23
    i32x8 gz8 = {};
    __builtin_amdgcn_tensor_load_to_lds(g0, g1, gz, gz, gz8, 0);
#else
    __builtin_amdgcn_tensor_load_to_lds(g0, g1, gz, gz, 0);
#endif
    __builtin_amdgcn_s_wait_tensorcnt(0);
  }
#else
  // fallback: plain interior copy (pairs; lo and lpad are even)
  for (int p = lane; p < ((hi - lo) >> 1); p += 32)
    *(unsigned*)&vtile[lpad + 2 * p] = *(const unsigned*)(vbf + lo + 2 * p);
#endif
  __syncthreads();

  const int m  = lane & 15;
  const int hi2 = lane >> 4;                    // lane half selects K sub-block
  const unsigned short* ap = vtile + 16 * m + 16 * hi2;
  const unsigned short* bp = bfrag + lane * 16;

  // dual accumulators break the WMMA D->C serial chain (5-slot bf16 hazard)
  v8f acc0 = {}, acc1 = {};
  for (int t = 0; t < NT - 1; t += 2) {
    __builtin_prefetch(bp + (t + 2) * 512, 0, 1);
    acc0 = conv_step(ap, bp, t, acc0);
    acc1 = conv_step(ap, bp, t + 1, acc1);
  }
  acc0 = conv_step(ap, bp, NT - 1, acc0);       // NT is odd
  v8f acc = acc0 + acc1;

  // D layout: VGPR r -> M = hi2*8 + r, N = lane&15
#pragma unroll
  for (int r = 0; r < 8; ++r)
    convout[j0 + (((hi2 << 3) + r) << 4) + (lane & 15)] = acc[r];
}

// ---------------- per-layer reductions: S1=sum tc^2, S2=sum tc*bk, S3=sum bk^2 ----------------
__global__ __launch_bounds__(256) void reduce3_kernel(
    const float* __restrict__ conv, const float* __restrict__ bk, int L,
    float* __restrict__ part) {
  __shared__ float sa[256], sb[256], sc[256];
  int tid = threadIdx.x;
  float a = 0.f, b = 0.f, c = 0.f;
  for (int j = blockIdx.x * 256 + tid; j < L; j += 256 * 256) {
    float y = bk[j];
    float t = (j < CONV_LEN) ? conv[j] : 0.f;
    a = fmaf(t, t, a); b = fmaf(t, y, b); c = fmaf(y, y, c);
  }
  sa[tid] = a; sb[tid] = b; sc[tid] = c;
  __syncthreads();
  for (int s = 128; s > 0; s >>= 1) {
    if (tid < s) { sa[tid] += sa[tid + s]; sb[tid] += sb[tid + s]; sc[tid] += sc[tid + s]; }
    __syncthreads();
  }
  if (tid == 0) {
    part[blockIdx.x] = sa[0];
    part[256 + blockIdx.x] = sb[0];
    part[512 + blockIdx.x] = sc[0];
  }
}

// ---------------- fold S1/S2/S3 into the two Mobius/expmap coefficients ----------------
__global__ __launch_bounds__(256) void finalize_layer_kernel(
    const float* __restrict__ part, float* __restrict__ scal) {
  __shared__ float sa[256], sb[256], sc[256];
  int tid = threadIdx.x;
  sa[tid] = part[tid]; sb[tid] = part[256 + tid]; sc[tid] = part[512 + tid];
  __syncthreads();
  for (int s = 128; s > 0; s >>= 1) {
    if (tid < s) { sa[tid] += sa[tid + s]; sb[tid] += sb[tid + s]; sc[tid] += sc[tid + s]; }
    __syncthreads();
  }
  if (tid == 0) {
    const float c = 1e-5f;
    const float sc2 = sqrtf(c);
    float S1 = sa[0], S2 = sb[0], S3 = sc[0];
    float n = fmaxf(sqrtf(S1), 1e-15f);
    float g = tanhf(sc2 * n) / (sc2 * n);            // expmap0 scalar factor
    float en = g * n;                                // norm after expmap0
    float maxnorm = (1.0f - 4e-3f) / sc2;
    float alpha = (en > maxnorm) ? (maxnorm / n) : g;  // proj folds into alpha
    float x2 = alpha * alpha * S1, y2 = S3, xy = alpha * S2;
    float base = 1.f + 2.f * c * xy;
    float A_ = base + c * y2;
    float B_ = 1.f - c * x2;
    float den = fmaxf(base + c * c * x2 * y2, 1e-15f);
    scal[0] = alpha * A_ / den;                      // multiplies total_conv
    scal[1] = B_ / den;                              // multiplies bk
  }
}

// ---------------- out = relu(cA*conv + cB*bk), plus bf16 copy for next conv ----------------
__global__ __launch_bounds__(256) void mobius_relu_kernel(
    const float* __restrict__ conv, const float* __restrict__ bk,
    const float* __restrict__ scal, int L, float* __restrict__ out,
    unsigned short* __restrict__ vbf) {
  int j = blockIdx.x * 256 + threadIdx.x;
  if (j >= L) return;
  float cA = scal[0], cB = scal[1];
  float t = (j < CONV_LEN) ? conv[j] : 0.f;
  float o = fmaxf(fmaf(cA, t, cB * bk[j]), 0.f);
  out[j] = o;
  if (vbf && j < INPUT_SIZE) vbf[j] = f2bf(o);
}

// ---------------- ||out||^2 and logmap0 scalar beta ----------------
__global__ __launch_bounds__(256) void reduce_sq_kernel(
    const float* __restrict__ x, int L, float* __restrict__ part) {
  __shared__ float s[256];
  int tid = threadIdx.x;
  float a = 0.f;
  for (int j = blockIdx.x * 256 + tid; j < L; j += 256 * 256) {
    float v = x[j];
    a = fmaf(v, v, a);
  }
  s[tid] = a;
  __syncthreads();
  for (int st = 128; st > 0; st >>= 1) {
    if (tid < st) s[tid] += s[tid + st];
    __syncthreads();
  }
  if (tid == 0) part[blockIdx.x] = s[0];
}

__global__ __launch_bounds__(256) void finalize_beta_kernel(
    const float* __restrict__ part, float* __restrict__ scal) {
  __shared__ float s[256];
  int tid = threadIdx.x;
  s[tid] = part[tid];
  __syncthreads();
  for (int st = 128; st > 0; st >>= 1) {
    if (tid < st) s[tid] += s[tid + st];
    __syncthreads();
  }
  if (tid == 0) {
    const float sc2 = sqrtf(1e-5f);
    float pn = fmaxf(sqrtf(s[0]), 1e-15f);
    float z = fminf(fmaxf(sc2 * pn, -1.f + 1e-7f), 1.f - 1e-7f);
    scal[2] = atanhf(z) / (sc2 * pn);                // logmap0 scalar factor
  }
}

// ---------------- HBM-bound GEMV: logits = beta * (trans_last @ out) ----------------
__global__ __launch_bounds__(256) void gemv_kernel(
    const float* __restrict__ trans, const float* __restrict__ x,
    const float* __restrict__ scal, float* __restrict__ logits) {
  const int row = blockIdx.x;
  const float4* tr = (const float4*)(trans + (size_t)row * OUT_MAX);
  const float4* xv = (const float4*)x;
  float a = 0.f;
  for (int k = threadIdx.x; k < OUT_MAX / 4; k += 256) {
    float4 t = tr[k];
    float4 v = xv[k];
    a = fmaf(t.x, v.x, fmaf(t.y, v.y, fmaf(t.z, v.z, fmaf(t.w, v.w, a))));
  }
  __shared__ float s[256];
  s[threadIdx.x] = a;
  __syncthreads();
  for (int st = 128; st > 0; st >>= 1) {
    if (threadIdx.x < st) s[threadIdx.x] += s[threadIdx.x + st];
    __syncthreads();
  }
  if (threadIdx.x == 0) logits[row] = scal[2] * s[0];
}

// ---------------- row-wise expmap0/proj then softmax over 512 classes ----------------
__global__ __launch_bounds__(512) void softmax_kernel(
    const float* __restrict__ logits, float* __restrict__ out) {
  __shared__ float s[512];
  int r = threadIdx.x;
  const float sc2 = sqrtf(1e-5f);
  float x = logits[r];
  float ax = fmaxf(fabsf(x), 1e-15f);
  float g = tanhf(sc2 * ax) / (sc2 * ax);
  float v = g * x;
  float n = fmaxf(fabsf(v), 1e-15f);
  float maxnorm = (1.f - 4e-3f) / sc2;
  if (n > maxnorm) v = v / n * maxnorm;
  s[r] = v;
  __syncthreads();
  for (int st = 256; st > 0; st >>= 1) {
    if (r < st) s[r] = fmaxf(s[r], s[r + st]);
    __syncthreads();
  }
  float mx = s[0];
  __syncthreads();
  float e = expf(v - mx);
  s[r] = e;
  __syncthreads();
  for (int st = 256; st > 0; st >>= 1) {
    if (r < st) s[r] += s[r + st];
    __syncthreads();
  }
  out[r] = e / s[0];
}

extern "C" void kernel_launch(void* const* d_in, const int* in_sizes, int n_in,
                              void* d_out, int out_size, void* d_ws, size_t ws_size,
                              hipStream_t stream) {
  (void)in_sizes; (void)n_in; (void)out_size; (void)ws_size;
  const float* hk = (const float*)d_in[0];
  const float* w[4]  = {(const float*)d_in[1], (const float*)d_in[3],
                        (const float*)d_in[5], (const float*)d_in[7]};
  const float* bk[4] = {(const float*)d_in[2], (const float*)d_in[4],
                        (const float*)d_in[6], (const float*)d_in[8]};
  const float* trans = (const float*)d_in[9];

  char* base = (char*)d_ws;
  size_t off = 0;
  auto alloc = [&](size_t bytes) -> void* {
    void* p = base + off;
    off = (off + bytes + 255) & ~(size_t)255;
    return p;
  };
  float* scal            = (float*)alloc(16 * sizeof(float));
  float* part            = (float*)alloc(1024 * sizeof(float));
  float* logits          = (float*)alloc(512 * sizeof(float));
  unsigned short* vbf    = (unsigned short*)alloc((size_t)INPUT_SIZE * 2);
  unsigned short* wrp    = (unsigned short*)alloc((size_t)WRP_LEN * 2);
  unsigned short* bfrag  = (unsigned short*)alloc((size_t)NT * 512 * 2);
  float* convb           = (float*)alloc((size_t)CONV_LEN * sizeof(float));
  float* outb            = (float*)alloc((size_t)OUT_MAX * sizeof(float));

  cvt_bf16_kernel<<<INPUT_SIZE / 256, 256, 0, stream>>>(hk, vbf, INPUT_SIZE);

  for (int i = 0; i < 4; ++i) {
    int L = INPUT_SIZE + (i + 1) * FILTER_LEN;
    build_wrp_kernel<<<(WRP_LEN + 255) / 256, 256, 0, stream>>>(w[i], wrp);
    build_bfrag_kernel<<<NT, 32, 0, stream>>>(wrp, bfrag);
    conv_wmma_kernel<<<CONV_LEN / 256, 32, 0, stream>>>(vbf, bfrag, convb);
    reduce3_kernel<<<256, 256, 0, stream>>>(convb, bk[i], L, part);
    finalize_layer_kernel<<<1, 256, 0, stream>>>(part, scal);
    mobius_relu_kernel<<<(L + 255) / 256, 256, 0, stream>>>(
        convb, bk[i], scal, L, outb, (i < 3) ? vbf : (unsigned short*)nullptr);
  }

  reduce_sq_kernel<<<256, 256, 0, stream>>>(outb, OUT_MAX, part);
  finalize_beta_kernel<<<1, 256, 0, stream>>>(part, scal);
  gemv_kernel<<<512, 256, 0, stream>>>(trans, outb, scal, logits);
  softmax_kernel<<<1, 512, 0, stream>>>(logits, (float*)d_out);
}
